// MultiHeadSelfAttention_29377576305097
// MI455X (gfx1250) — matrix operand order repaired and checked by reading, not compiled, not run on hardware
//
#include <hip/hip_runtime.h>
#include <hip/hip_bf16.h>

// MI455X / gfx1250 multi-head self-attention, bf16 WMMA (16x16x32, f32 acc).
// Data movement: async global->LDS (ASYNCcnt) in the GEMMs, TDM
// tensor_load_to_lds (TENSORcnt) for attention key tiles.
// B=4, L=2048, D=1024, H=16, HD=64.

typedef __bf16 bf16_t;
typedef __attribute__((ext_vector_type(16))) __bf16 v16bf;
typedef __attribute__((ext_vector_type(8)))  __bf16 v8bf;
typedef __attribute__((ext_vector_type(8)))  float  v8f;
typedef __attribute__((ext_vector_type(4)))  unsigned int v4u;
typedef __attribute__((ext_vector_type(8)))  int v8i;
typedef __attribute__((ext_vector_type(4)))  int v4i;

#define B_  4
#define L_  2048
#define D_  1024
#define H_  16
#define HD_ 64
#define BL_ (B_ * L_)   // 8192

// ---------------------------------------------------------------- helpers ---

__device__ __forceinline__ v8f v8f_zero() {
  v8f r;
#pragma unroll
  for (int i = 0; i < 8; ++i) r[i] = 0.0f;
  return r;
}

__device__ __forceinline__ v8f wmma_bf16(v16bf a, v16bf b, v8f c) {
  // D = A(16x32 bf16) * B(32x16 bf16) + C(16x16 f32)
  return __builtin_amdgcn_wmma_f32_16x16x32_bf16(
      /*neg_a=*/false, a, /*neg_b=*/false, b,
      /*c_mod=*/(short)0, c, /*reuse_a=*/false, /*reuse_b=*/false);
}

__device__ __forceinline__ v16bf pack16(v8bf lo, v8bf hi) {
  v16bf r;
#pragma unroll
  for (int i = 0; i < 8; ++i) { r[i] = lo[i]; r[8 + i] = hi[i]; }
  return r;
}

// A-matrix 16x32 bf16 fragment (row-major source):
//  lane<16 : row=lane,    K = {0..7, 16..23}
//  lane>=16: row=lane-16, K = {8..15, 24..31}
__device__ __forceinline__ v16bf load_a_frag(const bf16_t* row_base, int stride, int lane) {
  const bf16_t* p = row_base + (size_t)(lane & 15) * stride + ((lane >> 4) * 8);
  return pack16(*(const v8bf*)p, *(const v8bf*)(p + 16));
}

// B-matrix 32x16 bf16 fragment from a B^T-layout buffer bt[n][k]:
//  lane<16 : col=lane,    K = {0..7, 16..23}
//  lane>=16: col=lane-16, K = {8..15, 24..31}
__device__ __forceinline__ v16bf load_bt_frag(const bf16_t* bt_base, int stride, int lane) {
  const bf16_t* p = bt_base + (size_t)(lane & 15) * stride + ((lane >> 4) * 8);
  return pack16(*(const v8bf*)p, *(const v8bf*)(p + 16));
}

__device__ __forceinline__ v8f shfl_xor_v8(v8f v, int m) {
  v8f r;
#pragma unroll
  for (int i = 0; i < 8; ++i) r[i] = __shfl_xor(v[i], m, 32);
  return r;
}

// Low 32 bits of a flat pointer to __shared__ == LDS byte address (ISA 10.2).
__device__ __forceinline__ uint32_t lds_addr_of(const void* p) {
  return (uint32_t)(uintptr_t)p;
}

// 16-byte async copy global -> LDS (per-lane), tracked by ASYNCcnt.
__device__ __forceinline__ void async_copy_b128(uint32_t lds_dst, const void* src) {
  asm volatile("global_load_async_to_lds_b128 %0, %1, off"
               :: "v"(lds_dst), "v"((uint64_t)(uintptr_t)src)
               : "memory");
}

template <int N>
__device__ __forceinline__ void wait_asynccnt() {
  asm volatile("s_wait_asynccnt %0" :: "n"(N) : "memory");
}

// TDM: 1D tensor DMA global -> LDS, nelems bf16 elements (contiguous).
// D# per CDNA5 ISA 8.3-8.5: group0 {count=1, lds_addr, global_addr, type=2};
// group1 {wg_mask=0, data_size=1 (2B), tensor_dim0=tile_dim0=nelems,
// dim1 unused, stride0=nelems}; groups 2/3 zero (<=2D tensor).
__device__ __forceinline__ void tdm_load_1d(uint32_t lds_dst, const void* gsrc,
                                            uint32_t nelems) {
  const uint64_t ga = (uint64_t)(uintptr_t)gsrc;
  v4u g0;
  g0[0] = 1u;                                                  // count=1
  g0[1] = lds_dst;                                             // lds_addr
  g0[2] = (uint32_t)ga;                                        // global_addr lo
  g0[3] = (uint32_t)((ga >> 32) & 0x01FFFFFFu) | (2u << 30);   // addr hi | type=2
  v8i g1;
  g1[0] = (int)(1u << 16);           // wg_mask=0 | data_size=1 (2 bytes)
  g1[1] = (int)(nelems << 16);       // tensor_dim0[15:0]
  g1[2] = (int)(1u << 16);           // tensor_dim0[31:16]=0 | tensor_dim1=1
  g1[3] = (int)(nelems << 16);       // tensor_dim1 hi=0 | tile_dim0=nelems
  g1[4] = 0;                         // tile_dim1=0 (1D), tile_dim2=0
  g1[5] = (int)nelems;               // tensor_dim0_stride lo32
  g1[6] = 0;
  g1[7] = 0;
  v4i z4 = {0, 0, 0, 0};
#if defined(__clang_major__) && (__clang_major__ >= 23)
  v8i z8 = {0, 0, 0, 0, 0, 0, 0, 0};
  __builtin_amdgcn_tensor_load_to_lds(g0, g1, z4, z4, z8, 0);
#else
  __builtin_amdgcn_tensor_load_to_lds(g0, g1, z4, z4, 0);
#endif
}

// ------------------------------------------------------------ f32 -> bf16 ---

__global__ void f32_to_bf16(const float* __restrict__ in, bf16_t* __restrict__ out, int n) {
  int i = blockIdx.x * blockDim.x + threadIdx.x;
  if (i < n) out[i] = (bf16_t)in[i];
}

// ------------------------------------------------------------- WMMA GEMM ----
// C[M,N] = A[M,K] * B[K,N], bf16 inputs, f32 accumulate.
// Block tile 64x128, 8 waves (2x4), wave tile 32x32, K-step 32.
// A tiles: double-buffered async global->LDS copies (ASYNCcnt).
// B tiles: register-staged (prefetched one step ahead), transposed on store.
// One barrier per K-step; async for tile k+1 is issued after compute of tile
// k, so it only ever targets the buffer no wave can still be reading.
// MODE 0: QKV epilogue — scatter bf16 into Q/K/V [B*H, L, HD] buffers.
// MODE 1: proj epilogue — f32 + bias into d_out.

template <int MODE>
__global__ void __launch_bounds__(256)
gemm_bf16_wmma(const bf16_t* __restrict__ A, const bf16_t* __restrict__ Bm,
               int M, int N, int K,
               bf16_t* __restrict__ qo, bf16_t* __restrict__ ko, bf16_t* __restrict__ vo,
               float* __restrict__ co, const float* __restrict__ bias) {
  __shared__ bf16_t lA[2][64][32];    // 8 KB, row-major, double-buffered
  __shared__ bf16_t lBt[2][128][32];  // 16 KB, B transposed: lBt[buf][n][k]

  const int t = threadIdx.x;
  const int lane = t & 31, wave = t >> 5;
  const int wm = wave >> 2, wn = wave & 3;          // wave grid 2 x 4
  const int m0 = blockIdx.y * 64, n0 = blockIdx.x * 128;

  const int arow = t >> 2, acol = (t & 3) * 8;      // A stage: 1 x b128 / thread
  const int brow = t >> 3, bcol = (t & 7) * 16;     // B stage: 2 x b128 / thread

  const bf16_t* aSrc0 = A + (size_t)(m0 + arow) * K + acol;
  const bf16_t* bSrc0 = Bm + (size_t)brow * N + n0 + bcol;
  const uint32_t aDst[2] = { lds_addr_of(&lA[0][arow][acol]),
                             lds_addr_of(&lA[1][arow][acol]) };

  const int KT = K / 32;

  // preload tile 0
  async_copy_b128(aDst[0], aSrc0);
  v8bf bx0 = *(const v8bf*)bSrc0;
  v8bf bx1 = *(const v8bf*)(bSrc0 + 8);

  v8f acc[2][2];
#pragma unroll
  for (int i = 0; i < 2; ++i)
#pragma unroll
    for (int j = 0; j < 2; ++j) acc[i][j] = v8f_zero();

  for (int kt = 0; kt < KT; ++kt) {
    const int cur = kt & 1, nxt = cur ^ 1;

    // stage B tile kt (transpose from regs)
#pragma unroll
    for (int i = 0; i < 8; ++i) {
      lBt[cur][bcol + i][brow] = bx0[i];
      lBt[cur][bcol + 8 + i][brow] = bx1[i];
    }

    wait_asynccnt<0>();   // this wave's async A(kt) has landed
    __syncthreads();      // everyone's A(kt)/B(kt) staged; prev compute done

    v16bf a0 = load_a_frag(&lA[cur][wm * 32 + 0][0], 32, lane);
    v16bf a1 = load_a_frag(&lA[cur][wm * 32 + 16][0], 32, lane);
    v16bf b0 = load_bt_frag(&lBt[cur][wn * 32 + 0][0], 32, lane);
    v16bf b1 = load_bt_frag(&lBt[cur][wn * 32 + 16][0], 32, lane);
    acc[0][0] = wmma_bf16(a0, b0, acc[0][0]);
    acc[0][1] = wmma_bf16(a0, b1, acc[0][1]);
    acc[1][0] = wmma_bf16(a1, b0, acc[1][0]);
    acc[1][1] = wmma_bf16(a1, b1, acc[1][1]);

    // kick off tile kt+1 while others may still be computing tile kt
    if (kt + 1 < KT) {
      async_copy_b128(aDst[nxt], aSrc0 + (kt + 1) * 32);
      const bf16_t* bn = bSrc0 + (size_t)(kt + 1) * 32 * N;
      bx0 = *(const v8bf*)bn;
      bx1 = *(const v8bf*)(bn + 8);
    }
  }

  // C layout: VGPR r, lane<16 -> (M=r, N=lane); lane>=16 -> (M=r+8, N=lane-16)
#pragma unroll
  for (int i = 0; i < 2; ++i)
#pragma unroll
    for (int j = 0; j < 2; ++j) {
      const int mbase = m0 + wm * 32 + i * 16 + ((lane >> 4) * 8);
      const int nn = n0 + wn * 32 + j * 16 + (lane & 15);
#pragma unroll
      for (int r = 0; r < 8; ++r) {
        const int gm = mbase + r;
        const float v = acc[i][j][r];
        if (MODE == 0) {
          const int which = nn >> 10;          // 0=Q,1=K,2=V
          const int rem = nn & 1023;
          const int hh = rem >> 6, dd = rem & 63;
          const int bb = gm >> 11, ll = gm & 2047;  // L_ = 2048
          bf16_t* dst = (which == 0) ? qo : (which == 1) ? ko : vo;
          dst[(((size_t)(bb * H_ + hh)) * L_ + ll) * HD_ + dd] = (bf16_t)v;
        } else {
          co[(size_t)gm * N + nn] = v + bias[nn];
        }
      }
    }
}

// --------------------------------------------------- flash attention fwd ----
// One block per (b, h, 64 query rows); 4 waves x 16 rows. Online softmax.
// S = (Q K^T) * 0.125 ; O = softmax(S) V ; out bf16 in [B, L, D] layout.
// K tile (contiguous 8 KB) staged by the Tensor Data Mover, issued by wave 0
// (TDM ignores EXEC; descriptor is uniform -> SGPRs); completion via
// s_wait_tensorcnt before the consume barrier. V transposed manually.

__global__ void __launch_bounds__(128)
attn_fwd(const bf16_t* __restrict__ Qg, const bf16_t* __restrict__ Kg,
         const bf16_t* __restrict__ Vg, bf16_t* __restrict__ Og) {
  __shared__ bf16_t lK[64][64];        // key tile [key][d] == B^T for QK^T (8 KB)
  __shared__ bf16_t lVt[64][64];       // V^T tile [d][key] == B^T for P*V  (8 KB)
  __shared__ bf16_t lP[4][16][64];     // per-wave P scratch (C->A relayout) (8 KB)

  const int t = threadIdx.x, lane = t & 31, wave = t >> 5;
  const int h = blockIdx.y, b = blockIdx.z;
  const size_t base = ((size_t)(b * H_ + h)) * L_ * HD_;
  const int q0 = blockIdx.x * 64 + wave * 16;
  const int koff = (lane >> 4) * 8;

  // Q fragments stay in registers for the whole key loop (16 rows x 64 d)
  v16bf qf[2];
#pragma unroll
  for (int c = 0; c < 2; ++c) {
    const bf16_t* p = Qg + base + (size_t)(q0 + (lane & 15)) * HD_ + c * 32 + koff;
    qf[c] = pack16(*(const v8bf*)p, *(const v8bf*)(p + 16));
  }

  v8f m_run, l_run = v8f_zero(), o[4];
#pragma unroll
  for (int i = 0; i < 8; ++i) m_run[i] = -1e30f;
#pragma unroll
  for (int dt = 0; dt < 4; ++dt) o[dt] = v8f_zero();

  const int srow = t >> 1, sc0 = (t & 1) * 32;   // V staging map: 32 elems/thread

  for (int kb = 0; kb < L_; kb += 64) {
    __syncthreads();
    // TDM: whole 64x64 bf16 key tile is one contiguous 8 KB block
    if (wave == 0)
      tdm_load_1d(lds_addr_of(&lK[0][0]), Kg + base + (size_t)kb * HD_, 64 * HD_);

    {  // stage V transposed (overlaps the TDM transfer)
      const bf16_t* vs = Vg + base + (size_t)(kb + srow) * HD_ + sc0;
#pragma unroll
      for (int j = 0; j < 4; ++j) {
        v8bf vv = *(const v8bf*)(vs + j * 8);
#pragma unroll
        for (int i2 = 0; i2 < 8; ++i2) lVt[sc0 + j * 8 + i2][srow] = vv[i2];
      }
      if (kb + 64 < L_) {  // global_prefetch_b8 next key block
        __builtin_prefetch((const void*)(vs + (size_t)64 * HD_), 0, 1);
        __builtin_prefetch((const void*)(Kg + base + (size_t)(kb + 64) * HD_), 0, 1);
      }
    }
    if (wave == 0) __builtin_amdgcn_s_wait_tensorcnt(0);
    __syncthreads();

    // S = Q K^T * scale : 4 key sub-tiles x 2 d-chunks = 8 WMMA
    v8f s[4];
#pragma unroll
    for (int nt = 0; nt < 4; ++nt) {
      v8f a = v8f_zero();
#pragma unroll
      for (int c = 0; c < 2; ++c) {
        v16bf bfr = load_bt_frag(&lK[nt * 16][c * 32], 64, lane);
        a = wmma_bf16(qf[c], bfr, a);
      }
#pragma unroll
      for (int i = 0; i < 8; ++i) a[i] *= 0.125f;  // 1/sqrt(64)
      s[nt] = a;
    }

    // online softmax: rows live across 16-lane halves -> xor-shuffle reduce
    v8f mx = s[0];
#pragma unroll
    for (int nt = 1; nt < 4; ++nt)
#pragma unroll
      for (int i = 0; i < 8; ++i) mx[i] = fmaxf(mx[i], s[nt][i]);
#pragma unroll
    for (int msk = 1; msk <= 8; msk <<= 1) {
      v8f o2 = shfl_xor_v8(mx, msk);
#pragma unroll
      for (int i = 0; i < 8; ++i) mx[i] = fmaxf(mx[i], o2[i]);
    }
    v8f m_new, corr;
#pragma unroll
    for (int i = 0; i < 8; ++i) {
      m_new[i] = fmaxf(m_run[i], mx[i]);
      corr[i] = __expf(m_run[i] - m_new[i]);
    }
    v8f rs = v8f_zero();
#pragma unroll
    for (int nt = 0; nt < 4; ++nt)
#pragma unroll
      for (int i = 0; i < 8; ++i) {
        s[nt][i] = __expf(s[nt][i] - m_new[i]);
        rs[i] += s[nt][i];
      }
#pragma unroll
    for (int msk = 1; msk <= 8; msk <<= 1) {
      v8f o2 = shfl_xor_v8(rs, msk);
#pragma unroll
      for (int i = 0; i < 8; ++i) rs[i] += o2[i];
    }
#pragma unroll
    for (int i = 0; i < 8; ++i) l_run[i] = l_run[i] * corr[i] + rs[i];
    m_run = m_new;
#pragma unroll
    for (int dt = 0; dt < 4; ++dt)
#pragma unroll
      for (int i = 0; i < 8; ++i) o[dt][i] *= corr[i];

    // P: C-layout -> A-layout via wave-private LDS (in-order DS + dscnt wait)
#pragma unroll
    for (int nt = 0; nt < 4; ++nt)
#pragma unroll
      for (int r = 0; r < 8; ++r)
        lP[wave][r + ((lane >> 4) * 8)][nt * 16 + (lane & 15)] = (bf16_t)s[nt][r];
    asm volatile("s_wait_dscnt 0" ::: "memory");

    // O += P V : 2 key-chunks x 4 d-tiles = 8 WMMA
#pragma unroll
    for (int kc = 0; kc < 2; ++kc) {
      const bf16_t* pp = &lP[wave][lane & 15][kc * 32 + koff];
      v16bf pf = pack16(*(const v8bf*)pp, *(const v8bf*)(pp + 16));
#pragma unroll
      for (int dt = 0; dt < 4; ++dt) {
        v16bf vfr = load_bt_frag(&lVt[dt * 16][kc * 32], 64, lane);
        o[dt] = wmma_bf16(pf, vfr, o[dt]);
      }
    }
  }

  // normalize and store bf16 attention output in [B, L, D] (D = h*64+d)
  v8f inv;
#pragma unroll
  for (int i = 0; i < 8; ++i) inv[i] = 1.0f / l_run[i];
#pragma unroll
  for (int dt = 0; dt < 4; ++dt)
#pragma unroll
    for (int r = 0; r < 8; ++r) {
      const int q = q0 + r + ((lane >> 4) * 8);
      const int dcol = h * HD_ + dt * 16 + (lane & 15);
      Og[((size_t)b * L_ + q) * D_ + dcol] = (bf16_t)(o[dt][r] * inv[r]);
    }
}

// ---------------------------------------------------------------- launch ----

extern "C" void kernel_launch(void* const* d_in, const int* in_sizes, int n_in,
                              void* d_out, int out_size, void* d_ws, size_t ws_size,
                              hipStream_t stream) {
  const float* x      = (const float*)d_in[0];
  const float* w_qkv  = (const float*)d_in[1];
  const float* w_proj = (const float*)d_in[2];
  const float* b_proj = (const float*)d_in[3];
  float* out = (float*)d_out;
  char* ws = (char*)d_ws;

  const size_t nX  = (size_t)BL_ * D_;       // 8 Mi elems
  const size_t nWq = (size_t)D_ * 3 * D_;    // 3 Mi elems
  const size_t nWp = (size_t)D_ * D_;        // 1 Mi elems

  size_t off = 0;
  bf16_t* xb  = (bf16_t*)(ws + off); off += nX  * 2;
  bf16_t* wqb = (bf16_t*)(ws + off); off += nWq * 2;
  bf16_t* wpb = (bf16_t*)(ws + off); off += nWp * 2;
  bf16_t* Qb  = (bf16_t*)(ws + off); off += nX  * 2;   // [B*H, L, HD]
  bf16_t* Kb  = (bf16_t*)(ws + off); off += nX  * 2;
  bf16_t* Vb  = (bf16_t*)(ws + off); off += nX  * 2;
  bf16_t* Ab  = (bf16_t*)(ws + off); off += nX  * 2;   // attn out [B, L, D]

  f32_to_bf16<<<(int)((nX  + 255) / 256), 256, 0, stream>>>(x, xb, (int)nX);
  f32_to_bf16<<<(int)((nWq + 255) / 256), 256, 0, stream>>>(w_qkv, wqb, (int)nWq);
  f32_to_bf16<<<(int)((nWp + 255) / 256), 256, 0, stream>>>(w_proj, wpb, (int)nWp);

  gemm_bf16_wmma<0><<<dim3(3 * D_ / 128, BL_ / 64), 256, 0, stream>>>(
      xb, wqb, BL_, 3 * D_, D_, Qb, Kb, Vb, nullptr, nullptr);

  attn_fwd<<<dim3(L_ / 64, H_, B_), 128, 0, stream>>>(Qb, Kb, Vb, Ab);

  gemm_bf16_wmma<1><<<dim3(D_ / 128, BL_ / 64), 256, 0, stream>>>(
      Ab, wpb, BL_, D_, D_, nullptr, nullptr, nullptr, out, b_proj);
}
